// HypformerLinearAttention_45157286150963
// MI455X (gfx1250) — matrix operand order repaired and checked by reading, not compiled, not run on hardware
//
#include <hip/hip_runtime.h>
#include <math.h>

// ---------------------------------------------------------------------------
// Hypformer causal linear attention, chunked formulation for MI455X (gfx1250).
//
// Shapes: B=16, N=4096, D=128, M=256, DV=128. Causal scan re-expressed as
// chunked linear attention (chunk C=64) so every reduction is a WMMA GEMM:
//   phi   = exp(xn @ W^T - 0.5*||xn||^2) / sqrt(M)      (64x128)@(128x256)
//   A     = tril(phi_q @ phi_k^T)                        (64x256)@(256x64)
//   y     = (phi_q @ S_prev + A @ V) / max(den, eps)
//   den   = phi_q . z_prev + rowsum(A)
//   S    += phi_k^T @ V ;  z += colsum(phi_k)
// One workgroup per batch (causal chain serializes N), 16 wave32s.
// State S (256x128 f32) lives in per-wave WMMA accumulators (8 tiles/wave);
// a bf16 transposed image is dumped to LDS each chunk for the phi_q@S GEMM.
// HBM-bound (~135 MB @ 23.3 TB/s ~ 6us); v_wmma_f32_16x16x32_bf16 compute is
// far below that, so the design is a single pass over HBM with all
// intermediates in LDS/registers. exp via v_exp_f32 (TRANS32, co-executes
// with VALU); y-normalize via v_rcp_f32.
// ---------------------------------------------------------------------------

typedef __bf16 bf16;
typedef __attribute__((ext_vector_type(2)))  __bf16 bf16x2;
typedef __attribute__((ext_vector_type(8)))  __bf16 bf16x8;
typedef __attribute__((ext_vector_type(16))) __bf16 bf16x16;
typedef __attribute__((ext_vector_type(8)))  float  v8f;

#define BB      16
#define NN      4096
#define DD      128
#define MM      256
#define DVV     128
#define CHUNK   64
#define NCHUNK  (NN / CHUNK)
#define NTHREADS 512
#define XSCALE  0.2973017787506803f   /* 1 / 128^0.25 */
#define RSQRT_M 0.0625f               /* 1 / sqrt(256) */

// ---- LDS layout (bytes), 16B aligned everywhere --------------------------
#define OFF_W     0                       /* W bf16 row-major [256][128]  64KB */
#define OFF_PHIQ  65536                   /* phi_q bf16 [64][256]         32KB */
#define OFF_PHIK  98304                   /* phi_k bf16 [64][256]         32KB */
#define OFF_PHIKT 131072                  /* phi_k^T bf16 [256][64]       32KB */
#define OFF_SBFT  163840                  /* S^T bf16 [128][256]          64KB */
#define OFF_VT    229376                  /* V^T bf16 [128][64]           16KB */
#define OFF_AQK   245760                  /* masked A bf16 [64][64]        8KB */
#define OFF_XNQ   253952                  /* xn_q bf16 [64][128]  (aliases ybuf) */
#define OFF_XNK   270336                  /* xn_k bf16 [64][128]  (aliases ybuf) */
#define OFF_YBUF  253952                  /* y f32 [64][128]              32KB */
#define OFF_Z     286720                  /* z f32 [256] */
#define OFF_DEN   287744                  /* den f32 [64] */
#define OFF_SQQ   288000                  /* 0.5*||xn_q||^2 f32 [64] */
#define OFF_SQK   288256                  /* 0.5*||xn_k||^2 f32 [64] */
#define OFF_SSQ   288512                  /* sum(y^2) f32 [64] */
#define SMEM_BYTES 288768

static __device__ __forceinline__ bf16x16 cat8(bf16x8 lo, bf16x8 hi) {
  return __builtin_shufflevector(lo, hi, 0,1,2,3,4,5,6,7,8,9,10,11,12,13,14,15);
}

// A fragment, 16x32 bf16, source row-major [row][k], ld in elements.
// Lane l holds row (base_row + l%16); K pairs: [k0 + (l/16)*8 .. +7] then +16.
static __device__ __forceinline__ bf16x16
load_frag_a(const bf16* base, int ld, int row, int k0, int lane) {
  const bf16* p = base + row * ld + k0 + ((lane >> 4) << 3);
  bf16x8 lo = *(const bf16x8*)p;
  bf16x8 hi = *(const bf16x8*)(p + 16);
  return cat8(lo, hi);
}

// B fragment, 32x16 bf16, source stored K-contiguous per column:
// element (k, col) at base[col*ld + k]. Lane l holds col (base_col + l%16),
// K runs [k0 + (l/16)*8 .. +7] then +16 — two ds_load_b128 per lane.
static __device__ __forceinline__ bf16x16
load_frag_b(const bf16* base, int ld, int col, int k0, int lane) {
  const bf16* p = base + col * ld + k0 + ((lane >> 4) << 3);
  bf16x8 lo = *(const bf16x8*)p;
  bf16x8 hi = *(const bf16x8*)(p + 16);
  return cat8(lo, hi);
}

static __device__ __forceinline__ v8f
wmma_bf16(bf16x16 a, bf16x16 b, v8f c) {
  return __builtin_amdgcn_wmma_f32_16x16x32_bf16(
      false, a, false, b, (short)0, c, false, false);
}

__global__ __launch_bounds__(NTHREADS)
void HypformerLinearAttention_45157286150963_kernel(
    const float* __restrict__ Qg, const float* __restrict__ Kg,
    const float* __restrict__ Vg, const float* __restrict__ Wg,
    float* __restrict__ out) {
  extern __shared__ char smem[];
  bf16*  Wl    = (bf16*)(smem + OFF_W);
  bf16*  phiq  = (bf16*)(smem + OFF_PHIQ);
  bf16*  phik  = (bf16*)(smem + OFF_PHIK);
  bf16*  phikT = (bf16*)(smem + OFF_PHIKT);
  bf16*  SbfT  = (bf16*)(smem + OFF_SBFT);
  bf16*  VT    = (bf16*)(smem + OFF_VT);
  bf16*  Aqk   = (bf16*)(smem + OFF_AQK);
  bf16*  xnq   = (bf16*)(smem + OFF_XNQ);
  bf16*  xnk   = (bf16*)(smem + OFF_XNK);
  float* ybuf  = (float*)(smem + OFF_YBUF);
  float* zl    = (float*)(smem + OFF_Z);
  float* denl  = (float*)(smem + OFF_DEN);
  float* sqq   = (float*)(smem + OFF_SQQ);
  float* sqk   = (float*)(smem + OFF_SQK);
  float* ssq   = (float*)(smem + OFF_SSQ);

  const int tid  = threadIdx.x;
  const int lane = tid & 31;
  const int wave = tid >> 5;           // 16 waves
  const int rl16 = lane & 15;
  const int b    = blockIdx.x;

  // ---- one-time: W -> LDS bf16 (row-major [m][d] is B-operand K-contiguous),
  //      z = 0, S = 0 -------------------------------------------------------
  for (int e = tid; e < MM * DD; e += NTHREADS) Wl[e] = (bf16)Wg[e];
  if (tid < MM) zl[tid] = 0.0f;
  v8f Sacc[8];                         // wave owns S tiles (mt=wave, vt=0..7)
  #pragma unroll
  for (int t = 0; t < 8; ++t)
    #pragma unroll
    for (int g = 0; g < 8; ++g) Sacc[t][g] = 0.0f;
  __syncthreads();

  for (int c = 0; c < NCHUNK; ++c) {
    const int n0 = c * CHUNK;

    // -- stage 0: zero per-chunk reductions --------------------------------
    if (tid < CHUNK) { sqq[tid] = 0.0f; sqk[tid] = 0.0f; ssq[tid] = 0.0f; }
    __syncthreads();

    // -- stage 1: load chunk; xn (scaled, bf16), V^T, 0.5*||xn||^2 ---------
    {
      const int row = tid >> 3, seg = tid & 7;
      const size_t rb = ((size_t)(b * NN + n0 + row)) * (DD + 1) + 1 + seg * 16;
      float aq = 0.0f, ak = 0.0f;
      #pragma unroll
      for (int dd = 0; dd < 16; ++dd) {
        float q = Qg[rb + dd] * XSCALE;
        float k = Kg[rb + dd] * XSCALE;
        xnq[row * DD + seg * 16 + dd] = (bf16)q;
        xnk[row * DD + seg * 16 + dd] = (bf16)k;
        aq += q * q; ak += k * k;
      }
      atomicAdd(&sqq[row], 0.5f * aq);
      atomicAdd(&sqk[row], 0.5f * ak);
      for (int e = tid; e < CHUNK * DVV; e += NTHREADS) {
        int i = e >> 7, v = e & (DVV - 1);
        VT[v * CHUNK + i] = (bf16)Vg[((size_t)(b * NN + n0 + i)) * DVV + v];
      }
      if (c + 1 < NCHUNK) {            // gfx1250 global_prefetch_b8
        const size_t nb = ((size_t)(b * NN + n0 + CHUNK + row)) * (DD + 1);
        __builtin_prefetch(Qg + nb, 0, 1);
        __builtin_prefetch(Kg + nb, 0, 1);
        __builtin_prefetch(Vg + ((size_t)(b * NN + n0 + CHUNK + row)) * DVV, 0, 1);
      }
    }
    __syncthreads();

    // -- stage 2: feature maps  u = xn @ W^T  -> phi = exp(u - sq)/sqrt(M) -
    // 128 output tiles (2 mats x 4 ntiles x 16 mtiles), 8 per wave.
    #pragma unroll
    for (int t = 0; t < 8; ++t) {
      const int job = wave * 8 + t;
      const int matk = job >> 6, rem = job & 63;
      const int nt = rem >> 4, mt = rem & 15;
      const bf16* xs = matk ? xnk : xnq;
      const float* sqs = matk ? sqk : sqq;
      v8f acc = {0,0,0,0,0,0,0,0};
      #pragma unroll
      for (int kk = 0; kk < 4; ++kk)
        acc = wmma_bf16(load_frag_a(xs, DD, nt * 16 + rl16, kk * 32, lane),
                        load_frag_b(Wl, DD, mt * 16 + rl16, kk * 32, lane), acc);
      const int m = mt * 16 + rl16;
      if (matk) {
        // phi_k: row-major + packed-pair transposed image (n adjacent per g)
        #pragma unroll
        for (int g = 0; g < 8; g += 2) {
          int n = nt * 16 + g + ((lane >> 4) << 3);
          float p0 = __expf(acc[g]     - sqs[n])     * RSQRT_M;
          float p1 = __expf(acc[g + 1] - sqs[n + 1]) * RSQRT_M;
          phik[n * MM + m]       = (bf16)p0;
          phik[(n + 1) * MM + m] = (bf16)p1;
          bf16x2 pr = {(bf16)p0, (bf16)p1};
          *(bf16x2*)&phikT[m * CHUNK + n] = pr;   // ds_store_b32
        }
      } else {
        #pragma unroll
        for (int g = 0; g < 8; ++g) {
          int n = nt * 16 + g + ((lane >> 4) << 3);
          phiq[n * MM + m] = (bf16)(__expf(acc[g] - sqs[n]) * RSQRT_M);
        }
      }
    }
    __syncthreads();

    // -- stage 3a: den_part[n] = phi_q[n] . z_prev -------------------------
    if (tid < CHUNK) {
      float a = 0.0f;
      for (int m = 0; m < MM; ++m) a += (float)phiq[tid * MM + m] * zl[m];
      denl[tid] = a;
    }
    __syncthreads();

    // -- stage 3b: z += colsum(phi_k);  stage 4a: dump S^T (pre-update) ----
    if (tid < MM) {
      float s = 0.0f;
      for (int i = 0; i < CHUNK; ++i) s += (float)phik[i * MM + tid];
      zl[tid] += s;
    }
    #pragma unroll
    for (int t = 0; t < 8; ++t) {      // tile (mt=wave, vt=t); m adjacent per g
      #pragma unroll
      for (int g = 0; g < 8; g += 2) {
        int m = wave * 16 + g + ((lane >> 4) << 3);
        int v = t * 16 + rl16;
        bf16x2 pr = {(bf16)Sacc[t][g], (bf16)Sacc[t][g + 1]};
        *(bf16x2*)&SbfT[v * MM + m] = pr;         // ds_store_b32
      }
    }
    __syncthreads();

    // -- stage 4b: A = tril(phi_q @ phi_k^T); den += rowsum(A) -------------
    {
      const int nt = wave >> 2, jt = wave & 3;   // 16 tiles, one per wave
      v8f acc = {0,0,0,0,0,0,0,0};
      #pragma unroll
      for (int kk = 0; kk < 8; ++kk)
        acc = wmma_bf16(load_frag_a(phiq, MM, nt * 16 + rl16, kk * 32, lane),
                        load_frag_b(phik, MM, jt * 16 + rl16, kk * 32, lane), acc);
      #pragma unroll
      for (int g = 0; g < 8; ++g) {
        int n = nt * 16 + g + ((lane >> 4) << 3);
        int j = jt * 16 + rl16;
        float v = (j <= n) ? acc[g] : 0.0f;
        Aqk[n * CHUNK + j] = (bf16)v;
        if (j <= n) atomicAdd(&denl[n], v);
      }
    }
    __syncthreads();

    // -- stage 5: y = (phi_q @ S_prev + A @ V) * rcp(max(den, eps)) --------
    #pragma unroll
    for (int t = 0; t < 2; ++t) {      // 32 y tiles, 2 per wave
      const int tile = wave * 2 + t;
      const int nt = tile >> 3, vt = tile & 7;
      v8f acc = {0,0,0,0,0,0,0,0};
      #pragma unroll
      for (int kk = 0; kk < 8; ++kk)   // inter-chunk, K = M = 256
        acc = wmma_bf16(load_frag_a(phiq, MM, nt * 16 + rl16, kk * 32, lane),
                        load_frag_b(SbfT, MM, vt * 16 + rl16, kk * 32, lane), acc);
      #pragma unroll
      for (int kk = 0; kk < 2; ++kk)   // intra-chunk, K = C = 64
        acc = wmma_bf16(load_frag_a(Aqk, CHUNK, nt * 16 + rl16, kk * 32, lane),
                        load_frag_b(VT, CHUNK, vt * 16 + rl16, kk * 32, lane), acc);
      #pragma unroll
      for (int g = 0; g < 8; ++g) {
        int n = nt * 16 + g + ((lane >> 4) << 3);
        int v = vt * 16 + rl16;
        float r = __builtin_amdgcn_rcpf(fmaxf(denl[n], 1e-6f));  // v_rcp_f32
        ybuf[n * DVV + v] = acc[g] * r;
      }
    }
    // -- stage 6: S += phi_k^T @ V  (register accumulators, K = C = 64) ----
    #pragma unroll
    for (int t = 0; t < 8; ++t) {
      v8f acc = Sacc[t];
      #pragma unroll
      for (int kk = 0; kk < 2; ++kk)
        acc = wmma_bf16(load_frag_a(phikT, CHUNK, wave * 16 + rl16, kk * 32, lane),
                        load_frag_b(VT, CHUNK, t * 16 + rl16, kk * 32, lane), acc);
      Sacc[t] = acc;
    }
    __syncthreads();

    // -- stage 7: hyperboloid lift + store [x0, y] -------------------------
    {
      const int row = tid >> 3, seg = tid & 7;
      float ss = 0.0f;
      #pragma unroll
      for (int dd = 0; dd < 16; ++dd) {
        float y = ybuf[row * DVV + seg * 16 + dd];
        ss += y * y;
      }
      atomicAdd(&ssq[row], ss);
      __syncthreads();
      const size_t ob = ((size_t)(b * NN + n0 + row)) * (DVV + 1);
      if (seg == 0) out[ob] = sqrtf(1.0f + ssq[row]);
      #pragma unroll
      for (int dd = 0; dd < 16; ++dd)
        out[ob + 1 + seg * 16 + dd] = ybuf[row * DVV + seg * 16 + dd];
    }
    __syncthreads();   // ybuf aliases xn buffers of the next chunk
  }
}

extern "C" void kernel_launch(void* const* d_in, const int* in_sizes, int n_in,
                              void* d_out, int out_size, void* d_ws, size_t ws_size,
                              hipStream_t stream) {
  const float* Q  = (const float*)d_in[0];
  const float* K  = (const float*)d_in[1];
  const float* V  = (const float*)d_in[2];
  const float* W  = (const float*)d_in[3];
  float* out = (float*)d_out;
  (void)in_sizes; (void)n_in; (void)d_ws; (void)ws_size; (void)out_size;
  hipLaunchKernelGGL(HypformerLinearAttention_45157286150963_kernel,
                     dim3(BB), dim3(NTHREADS), SMEM_BYTES, stream,
                     Q, K, V, W, out);
}
